// SANet_74698071212166
// MI455X (gfx1250) — matrix operand order repaired and checked by reading, not compiled, not run on hardware
//
#include <hip/hip_runtime.h>
#include <hip/hip_bf16.h>
#include <math.h>

#define BATCH 4
#define CCH   512
#define NSP   4096   /* 64*64 spatial positions */

typedef __attribute__((ext_vector_type(16))) _Float16 v16h;
typedef __attribute__((ext_vector_type(8)))  _Float16 v8h;
typedef __attribute__((ext_vector_type(8)))  float    v8f;

// ---------------------------------------------------------------------------
// WMMA fragment helpers (CDNA5 16x16x32 f16, wave32, A/B in contiguous k order: k = 16*(lane/16) + s)
// ---------------------------------------------------------------------------

union frag_u { v16h v; struct { v8h lo; v8h hi; } s; };

// A fragment: row-major A[m][k], lda in elements. Lane L holds row M=L&15.
// Lanes 0-15: halves 0..15 = K(k0+0..15)
// Lanes 16-31: halves 0..15 = K(k0+16..31)
__device__ __forceinline__ v16h load_A_frag(const _Float16* A, int lda,
                                            int mtile, int k0, int lane) {
    int m   = mtile + (lane & 15);
    int klo = (lane < 16) ? 0 : 16;
    const _Float16* p = A + (size_t)m * lda + k0 + klo;
    frag_u u;
    u.s.lo = *(const v8h*)(p);        // 16B aligned (k0%32==0, klo%16==0)
    u.s.hi = *(const v8h*)(p + 8);
    return u.v;
}

// B fragment: operand stored K-contiguous as BT[n][k], ldb in elements.
// Lane L holds column N=L&15; lanes 0-15 K=k0..k0+15, lanes 16-31 K=k0+16..31.
__device__ __forceinline__ v16h load_B_frag(const _Float16* BT, int ldb,
                                            int ntile, int k0, int lane) {
    int n  = ntile + (lane & 15);
    int kb = (lane < 16) ? 0 : 16;
    return *(const v16h*)(BT + (size_t)n * ldb + k0 + kb);  // 32B contiguous
}

__device__ __forceinline__ v8f wmma_f16(v16h a, v16h b, v8f c) {
    // 8 args: (neg_a, A, neg_b, B, c_mod, C, reuse_a, reuse_b)
    return __builtin_amdgcn_wmma_f32_16x16x32_f16(false, a, false, b,
                                                  (short)0, c, false, false);
}

// ---------------------------------------------------------------------------
// Instance-norm statistics: mean + unbiased rstd per (b,c) over N=4096
// ---------------------------------------------------------------------------
__global__ void stats_kernel(const float* __restrict__ x,
                             float* __restrict__ mean,
                             float* __restrict__ rstd) {
    int bc = blockIdx.x;                       // b*C + c
    const float* p = x + (size_t)bc * NSP;
    float s = 0.f, s2 = 0.f;
    for (int i = threadIdx.x; i < NSP; i += blockDim.x) {
        float v = p[i];
        s += v; s2 += v * v;
    }
    __shared__ float shs[256];
    __shared__ float shq[256];
    shs[threadIdx.x] = s; shq[threadIdx.x] = s2;
    __syncthreads();
    for (int o = 128; o > 0; o >>= 1) {
        if (threadIdx.x < o) {
            shs[threadIdx.x] += shs[threadIdx.x + o];
            shq[threadIdx.x] += shq[threadIdx.x + o];
        }
        __syncthreads();
    }
    if (threadIdx.x == 0) {
        float m   = shs[0] / (float)NSP;
        float var = (shq[0] - (float)NSP * m * m) / (float)(NSP - 1); // ddof=1
        mean[bc] = m;
        rstd[bc] = rsqrtf(var + 1e-5f);
    }
}

// ---------------------------------------------------------------------------
// Normalize + transpose-pack one batch to f16, K(=channel)-contiguous.
// ---------------------------------------------------------------------------
__global__ void pack_kernel(const float* __restrict__ content,
                            const float* __restrict__ style,
                            const float* __restrict__ meanC, const float* __restrict__ rstdC,
                            const float* __restrict__ meanS, const float* __restrict__ rstdS,
                            _Float16* __restrict__ ncT,
                            _Float16* __restrict__ nsT,
                            _Float16* __restrict__ stT, int b) {
    int c = blockIdx.x;
    size_t base = ((size_t)b * CCH + c) * NSP;
    float mc = meanC[b * CCH + c], rc = rstdC[b * CCH + c];
    float ms = meanS[b * CCH + c], rs = rstdS[b * CCH + c];
    for (int n = threadIdx.x; n < NSP; n += blockDim.x) {
        float cv = content[base + n];
        float sv = style[base + n];
        ncT[(size_t)n * CCH + c] = (_Float16)((cv - mc) * rc);
        nsT[(size_t)n * CCH + c] = (_Float16)((sv - ms) * rs);
        stT[(size_t)n * CCH + c] = (_Float16)sv;
    }
}

// Convert a weight matrix f32 -> f16 (row-major, K-contiguous already).
__global__ void pack_w_kernel(const float* __restrict__ w,
                              _Float16* __restrict__ wh, int n) {
    int i = blockIdx.x * blockDim.x + threadIdx.x;
    if (i < n) wh[i] = (_Float16)w[i];
}

// ---------------------------------------------------------------------------
// Projection GEMM: D[m][n] = sum_k W[m][k] * XT[n][k] + bias[m]
//   M=C=512, Ncol=NSP=4096, K=C=512. One wave per 32x32 macro-tile (2x2 WMMA).
//   TO=1 -> store out[n*C+m] (qT/kT); TO=0 -> out[m*NSP+n] (v)
// ---------------------------------------------------------------------------
template <int TO>
__global__ __launch_bounds__(256, 2)
void gemm_proj(const _Float16* __restrict__ W,
               const float* __restrict__ bias,
               const _Float16* __restrict__ XT,
               _Float16* __restrict__ out) {
    int lane = threadIdx.x & 31;
    int wave = threadIdx.x >> 5;
    int tile = blockIdx.x * 8 + wave;          // 2048 macro tiles
    int nt = (tile & 127) * 32;                // 128 n-macro-tiles
    int mt = (tile >> 7) * 32;                 // 16 m-macro-tiles
    v8f acc[2][2] = {};
    for (int k0 = 0; k0 < CCH; k0 += 32) {
        v16h a[2], b[2];
#pragma unroll
        for (int i = 0; i < 2; ++i) a[i] = load_A_frag(W,  CCH, mt + 16 * i, k0, lane);
#pragma unroll
        for (int j = 0; j < 2; ++j) b[j] = load_B_frag(XT, CCH, nt + 16 * j, k0, lane);
#pragma unroll
        for (int i = 0; i < 2; ++i)
#pragma unroll
            for (int j = 0; j < 2; ++j)
                acc[i][j] = wmma_f16(a[i], b[j], acc[i][j]);
    }
    int cl  = lane & 15;
    int off = (lane < 16) ? 0 : 8;
#pragma unroll
    for (int i = 0; i < 2; ++i)
#pragma unroll
        for (int j = 0; j < 2; ++j) {
            int col = nt + 16 * j + cl;
#pragma unroll
            for (int r = 0; r < 8; ++r) {
                int row = mt + 16 * i + off + r;
                float dv = acc[i][j][r] + bias[row];
                if (TO) out[(size_t)col * CCH + row] = (_Float16)dv;
                else    out[(size_t)row * NSP + col] = (_Float16)dv;
            }
        }
}

// ---------------------------------------------------------------------------
// Logits GEMM: logits[n][m] = sum_c qT[n][c] * kT[m][c]   (f32 out)
//   64x64 macro-tile per wave: 4 A-frags + 4 B-frags -> 16 WMMAs per K-step.
//   launch_bounds(256,1): allow full 256-VGPR budget (128-reg accumulator).
// ---------------------------------------------------------------------------
__global__ __launch_bounds__(256, 1)
void gemm_logits(const _Float16* __restrict__ qT,
                 const _Float16* __restrict__ kT,
                 float* __restrict__ logits) {
    int lane = threadIdx.x & 31;
    int wave = threadIdx.x >> 5;
    int tile = blockIdx.x * 8 + wave;          // 4096 macro tiles
    int mt = (tile & 63) * 64;                 // style positions (cols)
    int nt = (tile >> 6) * 64;                 // content positions (rows)
    v8f acc[4][4] = {};
    for (int k0 = 0; k0 < CCH; k0 += 32) {
        v16h a[4], b[4];
#pragma unroll
        for (int i = 0; i < 4; ++i) a[i] = load_A_frag(qT, CCH, nt + 16 * i, k0, lane);
#pragma unroll
        for (int j = 0; j < 4; ++j) b[j] = load_B_frag(kT, CCH, mt + 16 * j, k0, lane);
#pragma unroll
        for (int i = 0; i < 4; ++i)
#pragma unroll
            for (int j = 0; j < 4; ++j)
                acc[i][j] = wmma_f16(a[i], b[j], acc[i][j]);
    }
    int cl  = lane & 15;
    int off = (lane < 16) ? 0 : 8;
#pragma unroll
    for (int i = 0; i < 4; ++i)
#pragma unroll
        for (int j = 0; j < 4; ++j) {
            int col = mt + 16 * j + cl;
#pragma unroll
            for (int r = 0; r < 8; ++r)
                logits[(size_t)(nt + 16 * i + off + r) * NSP + col] = acc[i][j][r];
        }
}

// ---------------------------------------------------------------------------
// Column softmax over n (rows) for each style position m, matching
// torch softmax(dim=1). f32 logits -> f16 attn, both [n][m] row-major.
// ---------------------------------------------------------------------------
__global__ void softmax_cols(const float* __restrict__ logits,
                             _Float16* __restrict__ attn) {
    int m = blockIdx.x * blockDim.x + threadIdx.x;   // 4096 columns
    float mx = -3.402823466e+38f;
    for (int n = 0; n < NSP; ++n)
        mx = fmaxf(mx, logits[(size_t)n * NSP + m]);
    float s = 0.f;
    for (int n = 0; n < NSP; ++n)
        s += __expf(logits[(size_t)n * NSP + m] - mx);
    float inv = 1.f / s;
    for (int n = 0; n < NSP; ++n)
        attn[(size_t)n * NSP + m] =
            (_Float16)(__expf(logits[(size_t)n * NSP + m] - mx) * inv);
}

// ---------------------------------------------------------------------------
// Output GEMM: out[c][n] = sum_m v[c][m] * attn[n][m]   (f32 out)
//   M=c (512), Ncol=n (4096), K=m (4096). 64x64 macro-tile per wave.
// ---------------------------------------------------------------------------
__global__ __launch_bounds__(256, 1)
void gemm_out(const _Float16* __restrict__ v,
              const _Float16* __restrict__ attn,
              float* __restrict__ out) {
    int lane = threadIdx.x & 31;
    int wave = threadIdx.x >> 5;
    int tile = blockIdx.x * 8 + wave;          // 512 macro tiles
    int nt = (tile & 63) * 64;                 // content positions (cols)
    int ct = (tile >> 6) * 64;                 // channels (rows)
    v8f acc[4][4] = {};
    for (int k0 = 0; k0 < NSP; k0 += 32) {
        if (k0 + 256 < NSP) {   // speculative prefetch -> global_prefetch_b8
            __builtin_prefetch(v    + (size_t)(ct + (lane & 15)) * NSP + k0 + 256, 0, 1);
            __builtin_prefetch(attn + (size_t)(nt + (lane & 15)) * NSP + k0 + 256, 0, 1);
        }
        v16h a[4], b[4];
#pragma unroll
        for (int i = 0; i < 4; ++i) a[i] = load_A_frag(v,    NSP, ct + 16 * i, k0, lane);
#pragma unroll
        for (int j = 0; j < 4; ++j) b[j] = load_B_frag(attn, NSP, nt + 16 * j, k0, lane);
#pragma unroll
        for (int i = 0; i < 4; ++i)
#pragma unroll
            for (int j = 0; j < 4; ++j)
                acc[i][j] = wmma_f16(a[i], b[j], acc[i][j]);
    }
    int cl  = lane & 15;
    int off = (lane < 16) ? 0 : 8;
#pragma unroll
    for (int i = 0; i < 4; ++i)
#pragma unroll
        for (int j = 0; j < 4; ++j) {
            int col = nt + 16 * j + cl;
#pragma unroll
            for (int r = 0; r < 8; ++r)
                out[(size_t)(ct + 16 * i + off + r) * NSP + col] = acc[i][j][r];
        }
}

// ---------------------------------------------------------------------------
// Host-side launch
// ---------------------------------------------------------------------------
extern "C" void kernel_launch(void* const* d_in, const int* in_sizes, int n_in,
                              void* d_out, int out_size, void* d_ws, size_t ws_size,
                              hipStream_t stream) {
    const float* content = (const float*)d_in[0];
    const float* style   = (const float*)d_in[1];
    const float* w1 = (const float*)d_in[2];
    const float* b1 = (const float*)d_in[3];
    const float* w2 = (const float*)d_in[4];
    const float* b2 = (const float*)d_in[5];
    const float* w3 = (const float*)d_in[6];
    const float* b3 = (const float*)d_in[7];
    float* outp = (float*)d_out;

    // workspace carve-up (256B aligned slices)
    char* p = (char*)d_ws;
    auto alloc = [&](size_t bytes) -> void* {
        void* r = (void*)p;
        p += (bytes + 255) & ~(size_t)255;
        return r;
    };
    float* meanC = (float*)alloc((size_t)BATCH * CCH * 4);
    float* rstdC = (float*)alloc((size_t)BATCH * CCH * 4);
    float* meanS = (float*)alloc((size_t)BATCH * CCH * 4);
    float* rstdS = (float*)alloc((size_t)BATCH * CCH * 4);
    _Float16* w1h = (_Float16*)alloc((size_t)CCH * CCH * 2);
    _Float16* w2h = (_Float16*)alloc((size_t)CCH * CCH * 2);
    _Float16* w3h = (_Float16*)alloc((size_t)CCH * CCH * 2);
    // per-batch reused buffers
    _Float16* ncT  = (_Float16*)alloc((size_t)NSP * CCH * 2);
    _Float16* nsT  = (_Float16*)alloc((size_t)NSP * CCH * 2);
    _Float16* stT  = (_Float16*)alloc((size_t)NSP * CCH * 2);
    _Float16* qT   = (_Float16*)alloc((size_t)NSP * CCH * 2);
    _Float16* kT   = (_Float16*)alloc((size_t)NSP * CCH * 2);
    _Float16* vv   = (_Float16*)alloc((size_t)CCH * NSP * 2);
    float*    logits = (float*)   alloc((size_t)NSP * NSP * 4);   // 64 MB
    _Float16* attn   = (_Float16*)alloc((size_t)NSP * NSP * 2);   // 32 MB

    // instance-norm stats (all batches at once)
    stats_kernel<<<BATCH * CCH, 256, 0, stream>>>(content, meanC, rstdC);
    stats_kernel<<<BATCH * CCH, 256, 0, stream>>>(style,   meanS, rstdS);

    // pack weights f32 -> f16 (row-major, K-contiguous)
    const int WN = CCH * CCH;
    pack_w_kernel<<<(WN + 255) / 256, 256, 0, stream>>>(w1, w1h, WN);
    pack_w_kernel<<<(WN + 255) / 256, 256, 0, stream>>>(w2, w2h, WN);
    pack_w_kernel<<<(WN + 255) / 256, 256, 0, stream>>>(w3, w3h, WN);

    for (int b = 0; b < BATCH; ++b) {
        pack_kernel<<<CCH, 256, 0, stream>>>(content, style,
                                             meanC, rstdC, meanS, rstdS,
                                             ncT, nsT, stT, b);

        // q = w1 * nc (store qT), k = w2 * ns (store kT), v = w3 * st (store v)
        gemm_proj<1><<<256, 256, 0, stream>>>(w1h, b1, ncT, qT);
        gemm_proj<1><<<256, 256, 0, stream>>>(w2h, b2, nsT, kT);
        gemm_proj<0><<<256, 256, 0, stream>>>(w3h, b3, stT, vv);

        // attention logits + column softmax + output GEMM
        gemm_logits<<<512, 256, 0, stream>>>(qT, kT, logits);
        softmax_cols<<<NSP / 256, 256, 0, stream>>>(logits, attn);
        gemm_out<<<64, 256, 0, stream>>>(vv, attn,
                                         outp + (size_t)b * CCH * NSP);
    }
}
